// MultiHeadLatentAttention_42795054137880
// MI455X (gfx1250) — matrix operand order repaired and checked by reading, not compiled, not run on hardware
//
#include <hip/hip_runtime.h>
#include <math.h>

#define B_ 2
#define T_ 2048
#define D_ 2048
#define H_ 16
#define DH_ 128
#define DKV_ 512
#define DQ_ 1024
#define DR_ 64
#define NT_ (B_*T_)           /* 4096 token rows */
#define DQK_ (DH_ + DR_)      /* 192 */
#define EPS_ 1e-6f
#define SCALE_ 0.07216878364870323f   /* 1/sqrt(192) */
#define ROPE_BASE_ 500000.0f

typedef __attribute__((ext_vector_type(16))) _Float16 v16h;
typedef __attribute__((ext_vector_type(8)))  _Float16 half8;
typedef __attribute__((ext_vector_type(8)))  float    v8f;

static __device__ inline v8f wmma16x16x32(v16h a, v16h b, v8f c) {
  // D = A(16x32 f16) * B(32x16 f16) + C(16x16 f32)
  return __builtin_amdgcn_wmma_f32_16x16x32_f16(false, a, false, b, (short)0, c,
                                                false, false);
}

// A-fragment (16x32 f16, MxK): lane m=lane&15, koff=(lane>>4)*8 holds
// K = koff..koff+7 (halves 0..7) and K = koff+16..koff+23 (halves 8..15).
static __device__ inline v16h load_frag_a(const _Float16* p) {
  half8 lo = *(const half8*)p;
  half8 hi = *(const half8*)(p + 16);
  v16h r;
#pragma unroll
  for (int i = 0; i < 8; ++i) { r[i] = lo[i]; r[8 + i] = hi[i]; }
  return r;
}

// B-fragment (32x16 f16, KxN): lane n=lane&15, koff=(lane>>4)*8 holds
// K = koff..koff+7 (halves 0..7) and K = koff+16..koff+23 (halves 8..15).
static __device__ inline v16h load_frag_b(const _Float16* p) {
  half8 lo = *(const half8*)p;
  half8 hi = *(const half8*)(p + 16);
  v16h r;
#pragma unroll
  for (int i = 0; i < 8; ++i) { r[i] = lo[i]; r[8 + i] = hi[i]; }
  return r;
}

// ---------------------------------------------------------------------------
// Generic WMMA GEMM: C(MxNpad) = A(MxK, f16 row-major) * BT(Npad x K, f16)^T
// Block = 128 threads = 4 waves; block tile 128x128, wave tile 64x64.
// ---------------------------------------------------------------------------
template <typename OutT>
__global__ __launch_bounds__(128) void gemm_wmma(
    const _Float16* __restrict__ A, const _Float16* __restrict__ BT,
    OutT* __restrict__ C, int K, int Npad) {
  const int lane = threadIdx.x & 31;
  const int wave = threadIdx.x >> 5;
  const int m0 = blockIdx.x * 128 + (wave & 1) * 64;
  const int n0 = blockIdx.y * 128 + (wave >> 1) * 64;
  const int mr   = lane & 15;
  const int hsel = lane >> 4;
  v8f acc[4][4] = {};
  for (int k0 = 0; k0 < K; k0 += 32) {
    v16h af[4], bf[4];
#pragma unroll
    for (int mi = 0; mi < 4; ++mi)
      af[mi] = load_frag_a(A + (size_t)(m0 + mi * 16 + mr) * K + k0 + hsel * 8);
#pragma unroll
    for (int ni = 0; ni < 4; ++ni)
      bf[ni] = load_frag_b(BT + (size_t)(n0 + ni * 16 + mr) * K + k0 + hsel * 8);
#pragma unroll
    for (int mi = 0; mi < 4; ++mi)
#pragma unroll
      for (int ni = 0; ni < 4; ++ni)
        acc[mi][ni] = wmma16x16x32(af[mi], bf[ni], acc[mi][ni]);
  }
  // C/D f32 layout: VGPR r -> row (lane<16 ? r : 8+r), col = lane&15
  const int rowb = hsel * 8;
#pragma unroll
  for (int mi = 0; mi < 4; ++mi)
#pragma unroll
    for (int ni = 0; ni < 4; ++ni)
#pragma unroll
      for (int r = 0; r < 8; ++r)
        C[(size_t)(m0 + mi * 16 + rowb + r) * Npad + n0 + ni * 16 + mr] =
            (OutT)acc[mi][ni][r];
}

// ---------------------------------------------------------------------------
// Elementwise helpers
// ---------------------------------------------------------------------------
__global__ __launch_bounds__(256) void cast_to_half(
    const float* __restrict__ in, _Float16* __restrict__ out, size_t n) {
  size_t i = (size_t)blockIdx.x * blockDim.x + threadIdx.x;
  if (i < n) out[i] = (_Float16)in[i];
}

// W (K x N, f32) -> WT (Npad x K, f16), zero-padded rows for n >= N
__global__ __launch_bounds__(256) void cast_transpose(
    const float* __restrict__ W, _Float16* __restrict__ WT, int K, int N,
    int Npad) {
  size_t idx = (size_t)blockIdx.x * blockDim.x + threadIdx.x;
  if (idx >= (size_t)Npad * K) return;
  int kk = (int)(idx % K);
  int n  = (int)(idx / K);
  WT[idx] = (n < N) ? (_Float16)W[(size_t)kk * N + n] : (_Float16)0.f;
}

__global__ __launch_bounds__(32) void rmsnorm_rows(
    const _Float16* __restrict__ in, const float* __restrict__ w,
    _Float16* __restrict__ out, int W) {
  const int row = blockIdx.x, lane = threadIdx.x;
  const _Float16* p = in + (size_t)row * W;
  float ss = 0.f;
  for (int j = lane; j < W; j += 32) { float v = (float)p[j]; ss += v * v; }
#pragma unroll
  for (int off = 16; off; off >>= 1) ss += __shfl_xor(ss, off, 32);
  const float sc = rsqrtf(ss / (float)W + EPS_);
  for (int j = lane; j < W; j += 32)
    out[(size_t)row * W + j] = (_Float16)((float)p[j] * sc * w[j]);
}

// content part: src (B*T, H*DH) -> dst (B,H,T,192), d in [0,128)
__global__ __launch_bounds__(256) void pack_content(
    const _Float16* __restrict__ src, _Float16* __restrict__ dst) {
  size_t idx = (size_t)blockIdx.x * blockDim.x + threadIdx.x;
  if (idx >= (size_t)B_ * H_ * T_ * DH_) return;
  int d = (int)(idx & (DH_ - 1));
  int t = (int)((idx >> 7) & (T_ - 1));
  int h = (int)((idx >> 18) & (H_ - 1));
  int b = (int)(idx >> 22);
  dst[((size_t)(b * H_ + h) * T_ + t) * DQK_ + d] =
      src[(size_t)(b * T_ + t) * (H_ * DH_) + h * DH_ + d];
}

static __device__ inline float rope_val(const _Float16* row, int t, int j) {
  int jj = j & 31;
  float invf = powf(ROPE_BASE_, -((float)(2 * jj)) / (float)DR_);
  float ang = (float)t * invf;
  float c = cosf(ang), s = sinf(ang);
  float v  = (float)row[j];
  float vr = (j < 32) ? -(float)row[j + 32] : (float)row[j - 32];
  return v * c + vr * s;
}

// rope part of q: qR (B*T, H*DR) -> dst (B,H,T,192) at d = 128+j
__global__ __launch_bounds__(256) void pack_rope_q(
    const _Float16* __restrict__ qR, _Float16* __restrict__ dst) {
  size_t idx = (size_t)blockIdx.x * blockDim.x + threadIdx.x;
  if (idx >= (size_t)B_ * H_ * T_ * DR_) return;
  int j = (int)(idx & (DR_ - 1));
  int t = (int)((idx >> 6) & (T_ - 1));
  int h = (int)((idx >> 17) & (H_ - 1));
  int b = (int)(idx >> 21);
  const _Float16* row = qR + (size_t)(b * T_ + t) * (H_ * DR_) + h * DR_;
  dst[((size_t)(b * H_ + h) * T_ + t) * DQK_ + DH_ + j] =
      (_Float16)rope_val(row, t, j);
}

// rope part of k (shared across heads): kR (B*T, 128pad; 64 valid)
__global__ __launch_bounds__(256) void pack_rope_k(
    const _Float16* __restrict__ kR, _Float16* __restrict__ dst) {
  size_t idx = (size_t)blockIdx.x * blockDim.x + threadIdx.x;
  if (idx >= (size_t)B_ * H_ * T_ * DR_) return;
  int j = (int)(idx & (DR_ - 1));
  int t = (int)((idx >> 6) & (T_ - 1));
  int h = (int)((idx >> 17) & (H_ - 1));
  int b = (int)(idx >> 21);
  const _Float16* row = kR + (size_t)(b * T_ + t) * 128;
  dst[((size_t)(b * H_ + h) * T_ + t) * DQK_ + DH_ + j] =
      (_Float16)rope_val(row, t, j);
}

// vC (B*T, H*DH) -> vT (B,H,DH,T)
__global__ __launch_bounds__(256) void pack_vt(
    const _Float16* __restrict__ vC, _Float16* __restrict__ vT) {
  size_t idx = (size_t)blockIdx.x * blockDim.x + threadIdx.x;
  if (idx >= (size_t)B_ * H_ * DH_ * T_) return;
  int t = (int)(idx & (T_ - 1));
  int d = (int)((idx >> 11) & (DH_ - 1));
  int h = (int)((idx >> 18) & (H_ - 1));
  int b = (int)(idx >> 22);
  vT[idx] = vC[(size_t)(b * T_ + t) * (H_ * DH_) + h * DH_ + d];
}

// ---------------------------------------------------------------------------
// Flash attention: one wave per (b, h, 16-row q tile). Online softmax over
// 32-key blocks; P transposed C-layout -> A-layout through a 1 KB LDS tile.
// ---------------------------------------------------------------------------
__global__ __launch_bounds__(32) void mla_attention(
    const _Float16* __restrict__ q, const _Float16* __restrict__ k,
    const _Float16* __restrict__ vT, _Float16* __restrict__ out) {
  __shared__ __align__(16) _Float16 Pl[16 * 32];
  const int lane = threadIdx.x;
  const int qt = blockIdx.x & 127;     // T/16 tiles
  const int bh = blockIdx.x >> 7;
  const int h  = bh & (H_ - 1);
  const int b  = bh >> 4;
  const int q0 = qt * 16;
  const int col  = lane & 15;
  const int hsel = lane >> 4;
  const int rowb = hsel * 8;

  // preload Q A-fragments for all 6 d-steps (reused over every key block)
  v16h qa[6];
  {
    const _Float16* qp = q + ((size_t)bh * T_ + q0 + col) * DQK_ + hsel * 8;
#pragma unroll
    for (int dsi = 0; dsi < 6; ++dsi) qa[dsi] = load_frag_a(qp + dsi * 32);
  }

  v8f acc[8] = {};
  float m[8], l[8];
#pragma unroll
  for (int r = 0; r < 8; ++r) { m[r] = -3.0e38f; l[r] = 0.f; }

  const int nkb = (q0 + 47) >> 5;  // causal: key blocks with kbase <= q0+15
  for (int kb = 0; kb < nkb; ++kb) {
    const int kbase = kb * 32;
    // scores: S(16x32) = Q(16x192) . K^T, two 16x16 C fragments
    v8f s0 = {}, s1 = {};
    const _Float16* kp0 = k + ((size_t)bh * T_ + kbase + col) * DQK_ + hsel * 8;
    const _Float16* kp1 = kp0 + 16 * DQK_;
#pragma unroll
    for (int dsi = 0; dsi < 6; ++dsi) {
      v16h b0 = load_frag_b(kp0 + dsi * 32);
      v16h b1 = load_frag_b(kp1 + dsi * 32);
      s0 = wmma16x16x32(qa[dsi], b0, s0);
      s1 = wmma16x16x32(qa[dsi], b1, s1);
    }
    // online softmax; row stats replicated across each 16-lane half
    float p0[8], p1[8], cf[8];
#pragma unroll
    for (int r = 0; r < 8; ++r) {
      const int qrow = q0 + rowb + r;
      float a0 = s0[r] * SCALE_; if (kbase + col      > qrow) a0 = -3.0e38f;
      float a1 = s1[r] * SCALE_; if (kbase + 16 + col > qrow) a1 = -3.0e38f;
      float mx = fmaxf(a0, a1);
      mx = fmaxf(mx, __shfl_xor(mx, 1, 32));
      mx = fmaxf(mx, __shfl_xor(mx, 2, 32));
      mx = fmaxf(mx, __shfl_xor(mx, 4, 32));
      mx = fmaxf(mx, __shfl_xor(mx, 8, 32));
      const float mn = fmaxf(m[r], mx);
      cf[r] = __expf(m[r] - mn);
      p0[r] = __expf(a0 - mn);
      p1[r] = __expf(a1 - mn);
      float sum = p0[r] + p1[r];
      sum += __shfl_xor(sum, 1, 32);
      sum += __shfl_xor(sum, 2, 32);
      sum += __shfl_xor(sum, 4, 32);
      sum += __shfl_xor(sum, 8, 32);
      l[r] = l[r] * cf[r] + sum;
      m[r] = mn;
    }
#pragma unroll
    for (int n = 0; n < 8; ++n)
#pragma unroll
      for (int r = 0; r < 8; ++r) acc[n][r] *= cf[r];
    // C-layout P -> LDS -> A-layout fragment
#pragma unroll
    for (int r = 0; r < 8; ++r) {
      Pl[(rowb + r) * 32 + col]      = (_Float16)p0[r];
      Pl[(rowb + r) * 32 + 16 + col] = (_Float16)p1[r];
    }
    __syncthreads();
    v16h pa = load_frag_a(Pl + (size_t)col * 32 + hsel * 8);
    __syncthreads();
    // acc(16x128) += P(16x32) . V(32x128), V read from (B,H,DH,T) transpose
    const _Float16* vp0 =
        vT + ((size_t)bh * DH_ + col) * T_ + kbase + hsel * 8;
#pragma unroll
    for (int n = 0; n < 8; ++n) {
      v16h vb = load_frag_b(vp0 + (size_t)n * 16 * T_);
      acc[n] = wmma16x16x32(pa, vb, acc[n]);
    }
  }
  // epilogue: normalize and scatter to (B, T, H*DH) f16
#pragma unroll
  for (int n = 0; n < 8; ++n)
#pragma unroll
    for (int r = 0; r < 8; ++r)
      out[((size_t)(b * T_) + q0 + rowb + r) * (size_t)(H_ * DH_) + h * DH_ +
          n * 16 + col] = (_Float16)(acc[n][r] / l[r]);
}

// ---------------------------------------------------------------------------
// Launcher
// ---------------------------------------------------------------------------
static inline unsigned cdiv(size_t a, unsigned b) {
  return (unsigned)((a + b - 1) / b);
}

extern "C" void kernel_launch(void* const* d_in, const int* in_sizes, int n_in,
                              void* d_out, int out_size, void* d_ws,
                              size_t ws_size, hipStream_t stream) {
  (void)in_sizes; (void)n_in; (void)out_size; (void)ws_size;
  const float* x     = (const float*)d_in[0];
  const float* W_DQ  = (const float*)d_in[1];
  const float* W_UQ  = (const float*)d_in[2];
  const float* W_QR  = (const float*)d_in[3];
  const float* W_DKV = (const float*)d_in[4];
  const float* W_UK  = (const float*)d_in[5];
  const float* W_UV  = (const float*)d_in[6];
  const float* W_KR  = (const float*)d_in[7];
  const float* W_O   = (const float*)d_in[8];
  const float* q_nw  = (const float*)d_in[9];
  const float* kv_nw = (const float*)d_in[10];
  float* outp = (float*)d_out;

  char* wp = (char*)d_ws;
  auto alloc = [&](size_t elems) -> _Float16* {
    _Float16* p = (_Float16*)wp;
    wp += ((elems * sizeof(_Float16)) + 255) & ~(size_t)255;
    return p;
  };
  _Float16* x16   = alloc((size_t)NT_ * D_);
  _Float16* WDQt  = alloc((size_t)DQ_ * D_);
  _Float16* WUQt  = alloc((size_t)(H_ * DH_) * DQ_);
  _Float16* WQRt  = alloc((size_t)(H_ * DR_) * DQ_);
  _Float16* WDKVt = alloc((size_t)DKV_ * D_);
  _Float16* WUKt  = alloc((size_t)(H_ * DH_) * DKV_);
  _Float16* WUVt  = alloc((size_t)(H_ * DH_) * DKV_);
  _Float16* WKRt  = alloc((size_t)128 * D_);          // DR=64 padded to 128
  _Float16* WOt   = alloc((size_t)D_ * (H_ * DH_));
  _Float16* tq    = alloc((size_t)NT_ * DQ_);
  _Float16* tkv   = alloc((size_t)NT_ * DKV_);
  _Float16* tkr   = alloc((size_t)NT_ * 128);
  _Float16* cQ    = alloc((size_t)NT_ * DQ_);
  _Float16* cKV   = alloc((size_t)NT_ * DKV_);
  _Float16* qC    = alloc((size_t)NT_ * (H_ * DH_));
  _Float16* qR    = alloc((size_t)NT_ * (H_ * DR_));
  _Float16* kC    = alloc((size_t)NT_ * (H_ * DH_));
  _Float16* vC    = alloc((size_t)NT_ * (H_ * DH_));
  _Float16* qbuf  = alloc((size_t)B_ * H_ * T_ * DQK_);
  _Float16* kbuf  = alloc((size_t)B_ * H_ * T_ * DQK_);
  _Float16* vTb   = alloc((size_t)B_ * H_ * DH_ * T_);
  _Float16* attn  = alloc((size_t)NT_ * (H_ * DH_));

  // 1) casts / weight transposes
  {
    size_t n = (size_t)NT_ * D_;
    cast_to_half<<<cdiv(n, 256), 256, 0, stream>>>(x, x16, n);
  }
  auto xt = [&](const float* W, _Float16* WT, int K, int N, int Npad) {
    size_t n = (size_t)Npad * K;
    cast_transpose<<<cdiv(n, 256), 256, 0, stream>>>(W, WT, K, N, Npad);
  };
  xt(W_DQ,  WDQt,  D_,   DQ_,       DQ_);
  xt(W_UQ,  WUQt,  DQ_,  H_ * DH_,  H_ * DH_);
  xt(W_QR,  WQRt,  DQ_,  H_ * DR_,  H_ * DR_);
  xt(W_DKV, WDKVt, D_,   DKV_,      DKV_);
  xt(W_UK,  WUKt,  DKV_, H_ * DH_,  H_ * DH_);
  xt(W_UV,  WUVt,  DKV_, H_ * DH_,  H_ * DH_);
  xt(W_KR,  WKRt,  D_,   DR_,       128);
  xt(W_O,   WOt,   H_ * DH_, D_,    D_);

  // 2) down-projections + rope key projection
  gemm_wmma<_Float16><<<dim3(NT_ / 128, DQ_ / 128), 128, 0, stream>>>(
      x16, WDQt, tq, D_, DQ_);
  gemm_wmma<_Float16><<<dim3(NT_ / 128, DKV_ / 128), 128, 0, stream>>>(
      x16, WDKVt, tkv, D_, DKV_);
  gemm_wmma<_Float16><<<dim3(NT_ / 128, 1), 128, 0, stream>>>(
      x16, WKRt, tkr, D_, 128);

  // 3) RMSNorm
  rmsnorm_rows<<<NT_, 32, 0, stream>>>(tq, q_nw, cQ, DQ_);
  rmsnorm_rows<<<NT_, 32, 0, stream>>>(tkv, kv_nw, cKV, DKV_);

  // 4) up-projections
  gemm_wmma<_Float16><<<dim3(NT_ / 128, (H_ * DH_) / 128), 128, 0, stream>>>(
      cQ, WUQt, qC, DQ_, H_ * DH_);
  gemm_wmma<_Float16><<<dim3(NT_ / 128, (H_ * DR_) / 128), 128, 0, stream>>>(
      cQ, WQRt, qR, DQ_, H_ * DR_);
  gemm_wmma<_Float16><<<dim3(NT_ / 128, (H_ * DH_) / 128), 128, 0, stream>>>(
      cKV, WUKt, kC, DKV_, H_ * DH_);
  gemm_wmma<_Float16><<<dim3(NT_ / 128, (H_ * DH_) / 128), 128, 0, stream>>>(
      cKV, WUVt, vC, DKV_, H_ * DH_);

  // 5) RoPE + layout packs
  {
    size_t n = (size_t)B_ * H_ * T_ * DH_;
    pack_content<<<cdiv(n, 256), 256, 0, stream>>>(qC, qbuf);
    pack_content<<<cdiv(n, 256), 256, 0, stream>>>(kC, kbuf);
    pack_vt<<<cdiv(n, 256), 256, 0, stream>>>(vC, vTb);
    size_t nr = (size_t)B_ * H_ * T_ * DR_;
    pack_rope_q<<<cdiv(nr, 256), 256, 0, stream>>>(qR, qbuf);
    pack_rope_k<<<cdiv(nr, 256), 256, 0, stream>>>(tkr, kbuf);
  }

  // 6) causal flash attention
  mla_attention<<<B_ * H_ * (T_ / 16), 32, 0, stream>>>(qbuf, kbuf, vTb, attn);

  // 7) output projection -> f32 d_out
  gemm_wmma<float><<<dim3(NT_ / 128, D_ / 128), 128, 0, stream>>>(
      attn, WOt, outp, H_ * DH_, D_);
}
